// FastMultiHeadAttention_54193897341248
// MI455X (gfx1250) — matrix operand order repaired and checked by reading, not compiled, not run on hardware
//
#include <hip/hip_runtime.h>
#include <hip/hip_bf16.h>

// ---------------------------------------------------------------------------
// CDNA5 (gfx1250) WMMA types, wave32
// ---------------------------------------------------------------------------
typedef __attribute__((ext_vector_type(16))) __bf16 v16bf;
typedef __attribute__((ext_vector_type(8)))  float  v8f;

#define EMBED 1024
#define HEADS 16
#define HDIM  64
#define SEQ   64
#define BATCH 1024
#define TILE  512           // elements per 16x32 (or 32x16) fragment tile

__device__ __forceinline__ unsigned short f2bfbits(float f) {
    union { float f; unsigned u; } v; v.f = f;
    unsigned u = v.u;
    unsigned r = u + 0x7FFFu + ((u >> 16) & 1u);   // round-to-nearest-even
    return (unsigned short)(r >> 16);
}

__device__ __forceinline__ v8f wmma_bf16(v16bf a, v16bf b, v8f c) {
    return __builtin_amdgcn_wmma_f32_16x16x32_bf16(
        false, a, false, b, (short)0, c, false, false);
}

// Fragment-major offsets within one 512-element tile (ISA 7.12.2 layouts).
// A tile is 16x32 (MxK): lane = (m&15) + 16*((k>>3)&1), idx = ((k>>4)<<3)|(k&7)
__device__ __forceinline__ int a_off(int m, int kloc) {
    return (((m & 15) + (((kloc >> 3) & 1) << 4)) << 4)
         + (((kloc >> 4) << 3) | (kloc & 7));
}
// B tile is 32x16 (KxN): lane = (n&15) + 16*((k>>3)&1), idx = ((k>>4)<<3)|(k&7)
__device__ __forceinline__ int b_off(int n, int kloc) {
    return (((n & 15) + (((kloc >> 3) & 1) << 4)) << 4)
         + (((kloc >> 4) << 3) | (kloc & 7));
}

// One fragment = 32 contiguous bytes per lane.
__device__ __forceinline__ v16bf load_frag(const unsigned short* p, int lane) {
    return *(const v16bf*)(p + lane * 16);
}

// ---------------------------------------------------------------------------
// Kernel 0a: weights fp32 -> bf16, fragment-tile-major.
//   wqf: [ntile(192)][kslab(32)][512]   (w_qkv as B operand, K=1024, N=3072)
//   wof: [ntile(64)][kslab(32)][512]    (w_out as B operand, K=1024, N=1024)
// ---------------------------------------------------------------------------
__global__ void cvt_weights(const float* __restrict__ wqkv,
                            const float* __restrict__ wout,
                            unsigned short* __restrict__ wqf,
                            unsigned short* __restrict__ wof) {
    int tid    = blockIdx.x * blockDim.x + threadIdx.x;
    int stride = gridDim.x * blockDim.x;
    const int NQ = EMBED * 3 * EMBED;
    const int NO = EMBED * EMBED;
    for (int i = tid; i < NQ; i += stride) {
        int k = i / (3 * EMBED), c = i % (3 * EMBED);
        wqf[((c >> 4) * 32 + (k >> 5)) * TILE + b_off(c, k & 31)] =
            f2bfbits(wqkv[i]);
    }
    for (int i = tid; i < NO; i += stride) {
        int k = i >> 10, n = i & (EMBED - 1);
        wof[((n >> 4) * 32 + (k >> 5)) * TILE + b_off(n, k & 31)] =
            f2bfbits(wout[i]);
    }
}

// ---------------------------------------------------------------------------
// Kernel 0b: x fp32 -> bf16, fragment-tile-major A operand.
//   xf: [b][kslab(32)][mtile(4)][512]
// ---------------------------------------------------------------------------
__global__ void cvt_x(const float* __restrict__ x,
                      unsigned short* __restrict__ xf) {
    size_t tid    = (size_t)blockIdx.x * blockDim.x + threadIdx.x;
    size_t stride = (size_t)gridDim.x * blockDim.x;
    const size_t N = (size_t)BATCH * SEQ * EMBED;       // 2^26
    for (size_t i = tid; i < N; i += stride) {
        int b = (int)(i >> 16);
        int m = (int)((i >> 10) & 63);
        int k = (int)(i & 1023);
        size_t dst = (((size_t)b * 32 + (k >> 5)) * 4 + (m >> 4)) * TILE
                   + a_off(m, k & 31);
        xf[dst] = f2bfbits(x[i]);
    }
}

// ---------------------------------------------------------------------------
// Kernel 1: fused per-(batch, head) QKV projection + attention.
// 128 threads = 4 waves.
// Phase 1 split: each wave owns 3 of the 12 qkv column tiles and ALL 4 row
// tiles -> per k-slab: 4 A-frag + 3 B-frag loads feeding 12 WMMAs (one wait
// point), instead of 13 loads / 12 waits.
// ---------------------------------------------------------------------------
__global__ void __launch_bounds__(128)
fused_attn(const unsigned short* __restrict__ xf,    // [b][ks][mtile][512]
           const unsigned short* __restrict__ wqf,   // [ntile(192)][ks][512]
           const float* __restrict__ bqkv,
           const float* __restrict__ pos_bias,       // (16, 64, 64)
           unsigned short* __restrict__ attnf)       // [b][ks(32)][mtile(4)][512]
{
    const int b = blockIdx.x / HEADS;
    const int h = blockIdx.x % HEADS;
    const int lane = threadIdx.x & 31;
    const int wave = threadIdx.x >> 5;
    const int lh   = (lane >> 4) << 3;   // 0 or 8 (C-layout row offset)
    const int n0   = lane & 15;

    // Fragment-major LDS: [tile][lane][16]
    __shared__ __attribute__((aligned(32))) unsigned short Qa[8 * TILE]; // A: [mtile(4)][ks(2)]
    __shared__ __attribute__((aligned(32))) unsigned short Kb[8 * TILE]; // B: [ntile(4)][ks(2)]
    __shared__ __attribute__((aligned(32))) unsigned short Vb[8 * TILE]; // B: [ntile(4)][ks(2)]
    __shared__ __attribute__((aligned(32))) unsigned short Pa[8 * TILE]; // A: [mtile(4)][ks(2)]

    // ---------------- Phase 1: QKV = x_b @ W_h + bias (64x192, K=1024) -----
    v8f acc[12];   // acc[mt*3 + j]
#pragma unroll
    for (int t = 0; t < 12; ++t) acc[t] = {};

    int tglob[3];
#pragma unroll
    for (int j = 0; j < 3; ++j) {
        int t = wave * 3 + j;                       // 0..11
        tglob[j] = (t >> 2) * 64 + h * 4 + (t & 3); // global wqf n-tile
    }

    const unsigned short* xb = xf + (size_t)b * 32 * 4 * TILE;
    for (int ks = 0; ks < 32; ++ks) {
        v16bf a[4], bb[3];
#pragma unroll
        for (int mt = 0; mt < 4; ++mt)
            a[mt] = load_frag(xb + (ks * 4 + mt) * TILE, lane);
#pragma unroll
        for (int j = 0; j < 3; ++j)
            bb[j] = load_frag(wqf + (tglob[j] * 32 + ks) * TILE, lane);
        if (ks + 1 < 32) {   // prefetch next k-slab (global_prefetch_b8)
            __builtin_prefetch(xb + ((ks + 1) * 4) * TILE + lane * 16, 0, 1);
            __builtin_prefetch(wqf + (tglob[0] * 32 + ks + 1) * TILE + lane * 16, 0, 1);
        }
#pragma unroll
        for (int mt = 0; mt < 4; ++mt)
#pragma unroll
            for (int j = 0; j < 3; ++j)
                acc[mt * 3 + j] = wmma_bf16(a[mt], bb[j], acc[mt * 3 + j]);
    }

    // Bias + scatter into fragment-major LDS (Q,P as A; K^T,V as B operands).
    // Wave-uniform tile->group mapping; all 12 tiles covered exactly once.
#pragma unroll
    for (int j = 0; j < 3; ++j) {
        int t    = wave * 3 + j;
        int grp  = t >> 2;                           // 0=q 1=k 2=v (uniform)
        int ncol = ((t & 3) << 4) + n0;              // 0..63 within head
        float bias = bqkv[grp * EMBED + h * HDIM + ncol];
#pragma unroll
        for (int mt = 0; mt < 4; ++mt) {
#pragma unroll
            for (int r = 0; r < 8; ++r) {
                int m = mt * 16 + r + lh;            // sequence row
                unsigned short bv = f2bfbits(acc[mt * 3 + j][r] + bias);
                if (grp == 0) {          // Q: A-layout
                    Qa[(mt * 2 + (ncol >> 5)) * TILE + a_off(m, ncol & 31)] = bv;
                } else if (grp == 1) {   // K^T: B-layout, n=seq row, k=ncol
                    Kb[(mt * 2 + (ncol >> 5)) * TILE + b_off(m, ncol & 31)] = bv;
                } else {                 // V: B-layout, n=ncol, k=seq row
                    Vb[((ncol >> 4) * 2 + (m >> 5)) * TILE + b_off(ncol, m & 31)] = bv;
                }
            }
        }
    }
    __syncthreads();

    // ---------------- Phase 2: S = Q K^T / 8 + pos_bias --------------------
    // Wave owns sequence row tile `wave` from here on.
    v8f sc[4];
#pragma unroll
    for (int t = 0; t < 4; ++t) sc[t] = {};
#pragma unroll
    for (int kk = 0; kk < 2; ++kk) {
        v16bf a = load_frag(Qa + (wave * 2 + kk) * TILE, lane);
#pragma unroll
        for (int t = 0; t < 4; ++t) {
            v16bf bb = load_frag(Kb + (t * 2 + kk) * TILE, lane);
            sc[t] = wmma_bf16(a, bb, sc[t]);
        }
    }
    const float* pb = pos_bias + h * SEQ * SEQ;
#pragma unroll
    for (int t = 0; t < 4; ++t) {
        int n = t * 16 + n0;
#pragma unroll
        for (int r = 0; r < 8; ++r) {
            int m = wave * 16 + r + lh;
            sc[t][r] = sc[t][r] * 0.125f + pb[m * SEQ + n];
        }
    }

    // ---------------- Phase 3: row softmax (row = 16 lanes of a half-wave) -
#pragma unroll
    for (int r = 0; r < 8; ++r) {
        float mx = fmaxf(fmaxf(sc[0][r], sc[1][r]), fmaxf(sc[2][r], sc[3][r]));
#pragma unroll
        for (int off = 1; off < 16; off <<= 1)
            mx = fmaxf(mx, __shfl_xor(mx, off, 32));
        float e0 = __expf(sc[0][r] - mx);
        float e1 = __expf(sc[1][r] - mx);
        float e2 = __expf(sc[2][r] - mx);
        float e3 = __expf(sc[3][r] - mx);
        float sum = e0 + e1 + e2 + e3;
#pragma unroll
        for (int off = 1; off < 16; off <<= 1)
            sum += __shfl_xor(sum, off, 32);
        float inv = 1.0f / sum;
        int m = wave * 16 + r + lh;
        Pa[(wave * 2 + 0) * TILE + a_off(m, (0  + n0) & 31)] = f2bfbits(e0 * inv);
        Pa[(wave * 2 + 0) * TILE + a_off(m, (16 + n0) & 31)] = f2bfbits(e1 * inv);
        Pa[(wave * 2 + 1) * TILE + a_off(m, (32 + n0) & 31)] = f2bfbits(e2 * inv);
        Pa[(wave * 2 + 1) * TILE + a_off(m, (48 + n0) & 31)] = f2bfbits(e3 * inv);
    }
    __syncthreads();

    // ---------------- Phase 4: O = P @ V -----------------------------------
    v8f oacc[4];
#pragma unroll
    for (int t = 0; t < 4; ++t) oacc[t] = {};
#pragma unroll
    for (int kk = 0; kk < 2; ++kk) {
        v16bf a = load_frag(Pa + (wave * 2 + kk) * TILE, lane);
#pragma unroll
        for (int t = 0; t < 4; ++t) {
            v16bf bb = load_frag(Vb + (t * 2 + kk) * TILE, lane);
            oacc[t] = wmma_bf16(a, bb, oacc[t]);
        }
    }

    // Swizzle O into A-fragment layout (reuse Qa; its reads finished before
    // the phase-3 barrier), then coalesced 32B global stores.
    unsigned short* Oa = Qa;   // [ks(2)][mtile(4)][512]
#pragma unroll
    for (int t = 0; t < 4; ++t) {
        int n = t * 16 + n0;                 // head-dim column 0..63
#pragma unroll
        for (int r = 0; r < 8; ++r) {
            int m = wave * 16 + r + lh;
            Oa[((n >> 5) * 4 + wave) * TILE + a_off(m, n & 31)] =
                f2bfbits(oacc[t][r]);
        }
    }
    __syncthreads();

    unsigned short* dst = attnf + (size_t)b * (32 * 4 * TILE) + h * (2 * 4 * TILE);
    for (int j = threadIdx.x; j < 256; j += 128) {
        *(v16bf*)(dst + j * 16) = *(const v16bf*)(Oa + j * 16);
    }
}

// ---------------------------------------------------------------------------
// Kernel 2: out = attn @ w_out + b_out  (65536x1024, K=1024) -> f32.
// LDS-free: both operands read as fragments straight from global.
// The 4x redundantly-loaded operand (wof) is the 2 MB L2-resident one.
// ---------------------------------------------------------------------------
__global__ void __launch_bounds__(128)
out_proj(const unsigned short* __restrict__ attnf,   // [b][ks][mtile][512]
         const unsigned short* __restrict__ wof,     // [ntile(64)][ks][512]
         const float* __restrict__ bout,
         float* __restrict__ out)
{
    const int nb = blockIdx.x;   // 0..15  (64 output cols)
    const int mb = blockIdx.y;   // 0..1023 (64 output rows)
    const int lane = threadIdx.x & 31;
    const int wave = threadIdx.x >> 5;

    v8f acc[4];
#pragma unroll
    for (int t = 0; t < 4; ++t) acc[t] = {};

    const unsigned short* arow = attnf + (size_t)mb * (32 * 4 * TILE);
    for (int ks = 0; ks < 32; ++ks) {
        v16bf a = load_frag(arow + (ks * 4 + wave) * TILE, lane);
        v16bf bb[4];
#pragma unroll
        for (int t = 0; t < 4; ++t)
            bb[t] = load_frag(wof + ((nb * 4 + t) * 32 + ks) * TILE, lane);
        if (ks + 1 < 32)
            __builtin_prefetch(arow + ((ks + 1) * 4 + wave) * TILE + lane * 16, 0, 1);
#pragma unroll
        for (int t = 0; t < 4; ++t)
            acc[t] = wmma_bf16(a, bb[t], acc[t]);
    }

#pragma unroll
    for (int t = 0; t < 4; ++t) {
        int n = nb * 64 + t * 16 + (lane & 15);
        float bias = bout[n];
#pragma unroll
        for (int r = 0; r < 8; ++r) {
            int m = mb * 64 + wave * 16 + r + ((lane >> 4) << 3);
            out[(size_t)m * EMBED + n] = acc[t][r] + bias;
        }
    }
}

// ---------------------------------------------------------------------------
// Host-side launcher
// ---------------------------------------------------------------------------
extern "C" void kernel_launch(void* const* d_in, const int* in_sizes, int n_in,
                              void* d_out, int out_size, void* d_ws, size_t ws_size,
                              hipStream_t stream) {
    const float* x        = (const float*)d_in[0];
    const float* w_qkv    = (const float*)d_in[1];
    const float* b_qkv    = (const float*)d_in[2];
    const float* pos_bias = (const float*)d_in[3];
    const float* w_out    = (const float*)d_in[4];
    const float* b_out    = (const float*)d_in[5];
    float* out = (float*)d_out;

    // Workspace (bf16 as ushort), all fragment-tile-major:
    //   wqf: 6 MB | wof: 2 MB | xf: 128 MB | attnf: 128 MB
    char* ws = (char*)d_ws;
    const size_t wq_bytes = (size_t)EMBED * 3 * EMBED * 2;
    const size_t wo_bytes = (size_t)EMBED * EMBED * 2;
    const size_t xf_bytes = (size_t)BATCH * SEQ * EMBED * 2;
    unsigned short* wqf   = (unsigned short*)ws;
    unsigned short* wof   = (unsigned short*)(ws + wq_bytes);
    unsigned short* xf    = (unsigned short*)(ws + wq_bytes + wo_bytes);
    unsigned short* attnf = (unsigned short*)(ws + wq_bytes + wo_bytes + xf_bytes);

    cvt_weights<<<1024, 256, 0, stream>>>(w_qkv, w_out, wqf, wof);
    cvt_x<<<8192, 256, 0, stream>>>(x, xf);

    fused_attn<<<BATCH * HEADS, 128, 0, stream>>>(xf, wqf, b_qkv, pos_bias, attnf);

    out_proj<<<dim3(16, BATCH), 128, 0, stream>>>(attnf, wof, b_out, out);
}
